// TokenPhraseLoss_3169685864484
// MI455X (gfx1250) — matrix operand order repaired and checked by reading, not compiled, not run on hardware
//
#include <hip/hip_runtime.h>
#include <math.h>

// Problem constants (reference: setup_inputs / N_HEADS / K1 / K2)
constexpr int B  = 8;
constexpr int L  = 512;
constexpr int D  = 768;
constexpr int H  = 12;
constexpr int DH = 64;     // D / H
constexpr int K1 = 20;
constexpr int K2 = 20;

typedef __attribute__((ext_vector_type(16))) __bf16 v16bf;
typedef __attribute__((ext_vector_type(8)))  float  v8f;

// ---------------------------------------------------------------------------
// Kernel 1: fused masked-score GEMM (bf16 WMMA) + pair loss + t-softmax
// One block = (b, head, 16-row i-tile). 256 threads = 8 waves.
// ---------------------------------------------------------------------------
__global__ void __launch_bounds__(256)
pair_kernel(const float* __restrict__ s_rep, const float* __restrict__ t_rep,
            const float* __restrict__ amask,
            float* __restrict__ g_score, float* __restrict__ loc,
            float* __restrict__ pair_sum) {
  const int blk  = blockIdx.x;
  const int it   = blk & 31;            // L/16 = 32 i-tiles
  const int h    = (blk >> 5) % H;
  const int b    = blk / (32 * H);
  const int tid  = threadIdx.x;
  const int lane = tid & 31;
  const int wid  = tid >> 5;
  const int i0   = it * 16;

  __shared__ float tileT[16][L];        // masked t scores, 32 KB
  __shared__ float maskSh[L];
  __shared__ float red[16][16];
  __shared__ float shPair;

  for (int x = tid; x < L; x += 256) maskSh[x] = amask[b * L + x];
  if (tid == 0) shPair = 0.0f;
  __syncthreads();

  const int half = lane >> 4;           // 0: lanes 0-15, 1: lanes 16-31
  const int l15  = lane & 15;
  const size_t baseRowA = ((size_t)b * L + (i0 + l15)) * D + h * DH;
  const float* sA = s_rep + baseRowA;
  const float* tA = t_rep + baseRowA;

  float pairLocal = 0.0f;
  for (int jt = wid; jt < L / 16; jt += 8) {
    const int j0 = jt * 16;
    const size_t baseRowB = ((size_t)b * L + (j0 + l15)) * D + h * DH;
    const float* sB = s_rep + baseRowB;
    const float* tB = t_rep + baseRowB;

    v8f cs = {};
    v8f ct = {};
#pragma unroll
    for (int ch = 0; ch < 2; ++ch) {    // K = 64 = 2 x 32
      const int kc = ch * 32;
      v16bf aS, aT, bS, bT;
#pragma unroll
      for (int e = 0; e < 16; ++e) {
        // 16-bit A 16x32 layout: lane-half -> K base 0/8; VGPR0-3 hold K 0..7, VGPR4-7 hold K 16..23
        const int ka = kc + half * 8 + e + ((e < 8) ? 0 : 8);
        // 16-bit B 32x16 layout: same K order as A (lane-half -> K base 0/8; elems 8-15 at K+16)
        const int kb = kc + half * 8 + e + ((e < 8) ? 0 : 8);
        aS[e] = (__bf16)sA[ka];
        aT[e] = (__bf16)tA[ka];
        bS[e] = (__bf16)sB[kb];
        bT[e] = (__bf16)tB[kb];
      }
      cs = __builtin_amdgcn_wmma_f32_16x16x32_bf16(false, aS, false, bS,
                                                   (short)0, cs, false, false);
      ct = __builtin_amdgcn_wmma_f32_16x16x32_bf16(false, aT, false, bT,
                                                   (short)0, ct, false, false);
    }

#pragma unroll
    for (int r = 0; r < 8; ++r) {
      const int M  = half * 8 + r;      // C/D layout: lanes 16-31 carry M = 8..15
      const int gi = i0 + M;
      const int gj = j0 + l15;
      const float mm = maskSh[gi] * maskSh[gj];
      const float sv = cs[r] * 0.125f * mm;   // 1/sqrt(64)
      const float tv = ct[r] * 0.125f * mm;
      const float dd = sv - tv;
      pairLocal += dd * dd;
      tileT[M][gj] = tv;
    }
  }
  atomicAdd(&shPair, pairLocal);
  __syncthreads();
  if (tid == 0) atomicAdd(pair_sum, shPair);

  // ---- softmax over each of the 16 rows (16 threads per row) ----
  const int r  = tid >> 4;
  const int c  = tid & 15;
  const int gi = i0 + r;
  const float mi = maskSh[gi];

  float mx = -INFINITY;
  for (int j = c; j < L; j += 16) {
    const float logit = tileT[r][j] + (1.0f - mi * maskSh[j]) * -10000.0f;
    mx = fmaxf(mx, logit);
  }
  red[r][c] = mx;
  __syncthreads();
  float rowMax = red[r][0];
#pragma unroll
  for (int kk = 1; kk < 16; ++kk) rowMax = fmaxf(rowMax, red[r][kk]);
  __syncthreads();

  float se = 0.0f;
  for (int j = c; j < L; j += 16) {
    const float logit = tileT[r][j] + (1.0f - mi * maskSh[j]) * -10000.0f;
    se += __expf(logit - rowMax);
  }
  red[r][c] = se;
  __syncthreads();
  float rowSum = 0.0f;
#pragma unroll
  for (int kk = 0; kk < 16; ++kk) rowSum += red[r][kk];
  const float inv = 1.0f / rowSum;

  for (int j = c; j < L; j += 16) {
    const float mj    = maskSh[j];
    const float logit = tileT[r][j] + (1.0f - mi * mj) * -10000.0f;
    const float att   = __expf(logit - rowMax) * inv * mi * mj;
    atomicAdd(&loc[((size_t)b * L + gi) * L + j], att);   // sum over heads
    atomicAdd(&g_score[b * L + j], att);                  // sum over heads & i
  }
}

// ---------------------------------------------------------------------------
// Kernel 2: top-K1 of g_score per batch (20 sequential parallel argmax passes)
// ---------------------------------------------------------------------------
__global__ void __launch_bounds__(256)
topk_g_kernel(const float* __restrict__ g_score, int* __restrict__ g_top) {
  const int b   = blockIdx.x;
  const int tid = threadIdx.x;
  __shared__ float vals[L];
  __shared__ float rv[256];
  __shared__ int   ri[256];

  for (int j = tid; j < L; j += 256) vals[j] = g_score[b * L + j];
  __syncthreads();

  for (int itr = 0; itr < K1; ++itr) {
    float bv = -INFINITY;
    int   bi = 0;
    for (int j = tid; j < L; j += 256) {
      const float v = vals[j];
      if (v > bv || (v == bv && j < bi)) { bv = v; bi = j; }
    }
    rv[tid] = bv; ri[tid] = bi;
    __syncthreads();
    for (int s = 128; s > 0; s >>= 1) {
      if (tid < s) {
        const float ov = rv[tid + s];
        const int   oi = ri[tid + s];
        if (ov > rv[tid] || (ov == rv[tid] && oi < ri[tid])) {
          rv[tid] = ov; ri[tid] = oi;
        }
      }
      __syncthreads();
    }
    if (tid == 0) { g_top[b * K1 + itr] = ri[0]; vals[ri[0]] = -INFINITY; }
    __syncthreads();
  }
}

// ---------------------------------------------------------------------------
// Kernel 3: top-K2 of loc[b, g_top[a], :] (self excluded)
// ---------------------------------------------------------------------------
__global__ void __launch_bounds__(256)
topk_l_kernel(const float* __restrict__ loc, const int* __restrict__ g_top,
              int* __restrict__ l_top) {
  const int a   = blockIdx.x;
  const int b   = blockIdx.y;
  const int tid = threadIdx.x;
  __shared__ float vals[L];
  __shared__ float rv[256];
  __shared__ int   ri[256];

  const int row = g_top[b * K1 + a];
  for (int j = tid; j < L; j += 256)
    vals[j] = (j == row) ? -INFINITY : loc[((size_t)b * L + row) * L + j];
  __syncthreads();

  for (int itr = 0; itr < K2; ++itr) {
    float bv = -INFINITY;
    int   bi = 0;
    for (int j = tid; j < L; j += 256) {
      const float v = vals[j];
      if (v > bv || (v == bv && j < bi)) { bv = v; bi = j; }
    }
    rv[tid] = bv; ri[tid] = bi;
    __syncthreads();
    for (int s = 128; s > 0; s >>= 1) {
      if (tid < s) {
        const float ov = rv[tid + s];
        const int   oi = ri[tid + s];
        if (ov > rv[tid] || (ov == rv[tid] && oi < ri[tid])) {
          rv[tid] = ov; ri[tid] = oi;
        }
      }
      __syncthreads();
    }
    if (tid == 0) { l_top[(b * K1 + a) * K2 + itr] = ri[0]; vals[ri[0]] = -INFINITY; }
    __syncthreads();
  }
}

// ---------------------------------------------------------------------------
// Kernel 4: triplet angles. One block = (anchor a, batch b, which in {s,t}).
// 20 normalized 768-dim difference vectors live in LDS (61.4 KB < 320 KB WGP).
// ---------------------------------------------------------------------------
__global__ void __launch_bounds__(256)
triplet_kernel(const float* __restrict__ s_rep, const float* __restrict__ t_rep,
               const float* __restrict__ amask,
               const int* __restrict__ g_top, const int* __restrict__ l_top,
               float* __restrict__ angle_s, float* __restrict__ angle_t,
               float* __restrict__ smask_sum) {
  const int a     = blockIdx.x;
  const int b     = blockIdx.y;
  const int which = blockIdx.z;   // 0 = s, 1 = t
  const int tid   = threadIdx.x;

  const float* hh = (which == 0) ? s_rep : t_rep;
  float* outAngle = (which == 0) ? angle_s : angle_t;

  __shared__ float nsd[K2][D];    // 61440 bytes
  __shared__ float red[256];
  __shared__ float shInv;
  __shared__ int   lidx[K2];
  __shared__ float lmask[K2];

  const int   g  = g_top[b * K1 + a];
  const float mg = amask[b * L + g];
  if (tid < K2) {
    const int l = l_top[(b * K1 + a) * K2 + tid];
    lidx[tid]  = l;
    lmask[tid] = amask[b * L + l];
  }
  __syncthreads();

  for (int j = 0; j < K2; ++j) {
    const int l = lidx[j];
    float ss = 0.0f;
    for (int d = tid; d < D; d += 256) {
      const float v = hh[((size_t)b * L + g) * D + d] -
                      hh[((size_t)b * L + l) * D + d];
      nsd[j][d] = v;
      ss += v * v;
    }
    red[tid] = ss;
    __syncthreads();
    for (int s = 128; s > 0; s >>= 1) {
      if (tid < s) red[tid] += red[tid + s];
      __syncthreads();
    }
    if (tid == 0) shInv = 1.0f / fmaxf(sqrtf(red[0]), 1e-12f);
    __syncthreads();
    const float inv = shInv;
    for (int d = tid; d < D; d += 256) nsd[j][d] *= inv;
    __syncthreads();
  }

  float smaskLocal = 0.0f;
  for (int p = tid; p < K2 * K2; p += 256) {
    const int j = p / K2, k = p % K2;
    float dot = 0.0f;
    for (int d = 0; d < D; ++d) dot += nsd[j][d] * nsd[k][d];
    const float me = (mg + lmask[j]) * (mg + lmask[k]);
    const bool on = (me == 4.0f) && (dot != 0.0f) && (j != k);
    outAngle[(((size_t)b * K1 + a) * K2 + j) * K2 + k] = on ? dot : 0.0f;
    if (which == 0 && on) smaskLocal += 1.0f;
  }
  if (which == 0) {
    red[tid] = smaskLocal;
    __syncthreads();
    for (int s = 128; s > 0; s >>= 1) {
      if (tid < s) red[tid] += red[tid + s];
      __syncthreads();
    }
    if (tid == 0) atomicAdd(smask_sum, red[0]);
  }
}

// ---------------------------------------------------------------------------
// Kernel 5: assemble scalar loss
// ---------------------------------------------------------------------------
__global__ void __launch_bounds__(256)
finalize_kernel(const float* __restrict__ amask, const float* __restrict__ pair_sum,
                const float* __restrict__ smask_sum,
                const float* __restrict__ angle_s, const float* __restrict__ angle_t,
                float* __restrict__ out) {
  const int tid = threadIdx.x;
  __shared__ float red[256];
  __shared__ float mextSum;

  float part = 0.0f;
  if (tid < B) {
    float s = 0.0f;
    for (int j = 0; j < L; ++j) s += amask[tid * L + j];
    part = s * s;                 // sum(mext) = sum_b (row-sum)^2
  }
  red[tid] = part;
  __syncthreads();
  for (int s = 128; s > 0; s >>= 1) {
    if (tid < s) red[tid] += red[tid + s];
    __syncthreads();
  }
  if (tid == 0) mextSum = red[0];
  __syncthreads();

  float sl = 0.0f;
  const int N = B * K1 * K2 * K2;
  for (int i = tid; i < N; i += 256) {
    const float d  = angle_s[i] - angle_t[i];
    const float ad = fabsf(d);
    sl += (ad < 1.0f) ? 0.5f * d * d : (ad - 0.5f);
  }
  red[tid] = sl;
  __syncthreads();
  for (int s = 128; s > 0; s >>= 1) {
    if (tid < s) red[tid] += red[tid + s];
    __syncthreads();
  }
  if (tid == 0) {
    const float loss_pair = pair_sum[0] / ((float)H * mextSum);
    const float loss_trip = red[0] / smask_sum[0];
    out[0] = loss_pair + loss_trip;
  }
}

// ---------------------------------------------------------------------------
extern "C" void kernel_launch(void* const* d_in, const int* in_sizes, int n_in,
                              void* d_out, int out_size, void* d_ws, size_t ws_size,
                              hipStream_t stream) {
  (void)in_sizes; (void)n_in; (void)out_size; (void)ws_size;
  const float* s_rep = (const float*)d_in[0];
  const float* t_rep = (const float*)d_in[1];
  const float* amask = (const float*)d_in[2];

  // Workspace layout (floats):
  float*  ws        = (float*)d_ws;
  float*  pair_sum  = ws;                                   // 1
  float*  smask_sum = ws + 1;                               // 1 (+2 pad)
  float*  g_score   = ws + 4;                               // B*L
  float*  loc       = g_score + B * L;                      // B*L*L
  const size_t locN = (size_t)B * L * L;
  int*    g_top     = (int*)(loc + locN);                   // B*K1
  int*    l_top     = g_top + B * K1;                       // B*K1*K2
  float*  angle_s   = (float*)(l_top + B * K1 * K2);        // B*K1*K2*K2
  float*  angle_t   = angle_s + (size_t)B * K1 * K2 * K2;   // B*K1*K2*K2

  // Zero the accumulated region (pair_sum, smask_sum, g_score, loc).
  const size_t zeroFloats = 4 + (size_t)B * L + locN;
  hipMemsetAsync(d_ws, 0, zeroFloats * sizeof(float), stream);

  pair_kernel<<<dim3(B * H * (L / 16)), 256, 0, stream>>>(
      s_rep, t_rep, amask, g_score, loc, pair_sum);
  topk_g_kernel<<<dim3(B), 256, 0, stream>>>(g_score, g_top);
  topk_l_kernel<<<dim3(K1, B), 256, 0, stream>>>(loc, g_top, l_top);
  triplet_kernel<<<dim3(K1, B, 2), 256, 0, stream>>>(
      s_rep, t_rep, amask, g_top, l_top, angle_s, angle_t, smask_sum);
  finalize_kernel<<<1, 256, 0, stream>>>(
      amask, pair_sum, smask_sum, angle_s, angle_t, (float*)d_out);
}
